// LSTMCell_25245817766082
// MI455X (gfx1250) — matrix operand order repaired and checked by reading, not compiled, not run on hardware
//
#include <hip/hip_runtime.h>
#include <cstdint>

// ---------------------------------------------------------------------------
// LSTM cell, fused 2-GEMM + elementwise, CDNA5 (gfx1250) WMMA bf16x3 path.
//   N = 64*4096 rows, F = 64, H = 128, K = F+H = 192, G = 4H = 512 cols.
// Precision: fp32 split into bf16 hi (truncate) + bf16 lo (truncated
//   residual); D = Ahi*Bhi + Ahi*Blo + Alo*Bhi (fp32 accumulate) ~= fp32.
// Weight staging global->LDS uses the Tensor Data Mover (TDM).
// ---------------------------------------------------------------------------

#define NROWS    (64 * 4096)           // 262144
#define ROWBLK   128                   // rows per WG iteration (8 waves x 16)
#define NRB      (NROWS / ROWBLK)      // 2048 row blocks
#define KT_N     6                     // 192 / 32 k-tiles
#define GRP_DW   12288                 // dwords per (hi or lo) plane per group
#define GRP_STRIDE (2 * GRP_DW)        // hi+lo contiguous per group (24576 dw)

typedef __attribute__((ext_vector_type(16))) __bf16 v16bf;
typedef __attribute__((ext_vector_type(8)))  float  v8f;
typedef unsigned int u32x4 __attribute__((ext_vector_type(4)));
typedef int          i32x8 __attribute__((ext_vector_type(8)));
typedef int          i32x4 __attribute__((ext_vector_type(4)));

union BReg {
    uint32_t u[8];
    uint4    q[2];
    v16bf    v;
};

__device__ __forceinline__ float fast_sigmoid(float x) {
    return __builtin_amdgcn_rcpf(1.0f + __expf(-x));
}
__device__ __forceinline__ float fast_tanh(float x) {
    // 1 - 2/(e^{2x}+1): branch-free, correct limits at +-inf.
    float e = __expf(2.0f * x);
    return 1.0f - 2.0f * __builtin_amdgcn_rcpf(e + 1.0f);
}

// Pack the two high halves of (b0, b1) into one dword: [15:0]=b0>>16,
// [31:16]=b1>>16.  Single v_perm_b32.
__device__ __forceinline__ uint32_t pack_hi16(uint32_t b0, uint32_t b1) {
    return __builtin_amdgcn_perm(b1, b0, 0x07060302u);
}

// Convert one lane's 16 fp32 A elements (two aligned 8-float runs) into the
// packed bf16 hi/lo register image required by V_WMMA_F32_16X16X32_BF16.
__device__ __forceinline__ void cvt_row_bf16x2(const float* __restrict__ src,
                                               BReg& hi, BReg& lo) {
    float4 a0 = *(const float4*)(src + 0);
    float4 a1 = *(const float4*)(src + 4);
    float4 a2 = *(const float4*)(src + 16);
    float4 a3 = *(const float4*)(src + 20);
    float v[16] = {a0.x, a0.y, a0.z, a0.w, a1.x, a1.y, a1.z, a1.w,
                   a2.x, a2.y, a2.z, a2.w, a3.x, a3.y, a3.z, a3.w};
#pragma unroll
    for (int j = 0; j < 8; ++j) {
        float f0 = v[2 * j], f1 = v[2 * j + 1];
        uint32_t b0 = __float_as_uint(f0);
        uint32_t b1 = __float_as_uint(f1);
        hi.u[j] = pack_hi16(b0, b1);                        // truncate -> bf16 hi
        float l0 = f0 - __uint_as_float(b0 & 0xFFFF0000u);  // exact residual
        float l1 = f1 - __uint_as_float(b1 & 0xFFFF0000u);
        lo.u[j] = pack_hi16(__float_as_uint(l0), __float_as_uint(l1));
    }
}

// ---------------------------------------------------------------------------
// Prep kernel: repack concatenated weights [192 x 512] into the per-lane
// B-operand layout of V_WMMA_F32_16X16X32_BF16, with columns permuted so a
// packed unit-group g2 holds (gate i|f|c|o) x 32 hidden units:
//   packed col: g2 in 0..3, nt in 0..7, c in 0..15
//   gate = nt>>1,  unit u = 32*g2 + 16*(nt&1) + c
// B tile (32x16) per-lane image: lane holds col (lane&15), K = kt*32 +
//   8*(lane>>4) + 16*(j>>2) + 2*(j&3) + p, dword j packs (K even -> [15:0], odd -> [31:16]).
// ws layout: per group g2 one contiguous 96 KB block: [hi 48 KB | lo 48 KB],
// tile order (kt*8 + nt) so the main kernel's per-kt DS offsets all fit the
// 16-bit immediate field. Staged with a single TDM transfer.
// ---------------------------------------------------------------------------
__global__ void lstm_pack_weights(
    const float* __restrict__ Wxi, const float* __restrict__ Whi_,
    const float* __restrict__ Wxf, const float* __restrict__ Whf,
    const float* __restrict__ Wxc, const float* __restrict__ Whc,
    const float* __restrict__ Wxo, const float* __restrict__ Who,
    uint32_t* __restrict__ pack) {
    int tid = blockIdx.x * blockDim.x + threadIdx.x;
    if (tid >= 4 * GRP_DW) return;
    int t = tid;
    int j    = t & 7;  t >>= 3;
    int lane = t & 31; t >>= 5;
    int kt   = t % 6;  t /= 6;
    int nt   = t & 7;
    int g2   = t >> 3;
    int c    = lane & 15;
    int u    = 32 * g2 + 16 * (nt & 1) + c;
    int gate = nt >> 1;
    const float* Wx[4] = {Wxi, Wxf, Wxc, Wxo};
    const float* Wh[4] = {Whi_, Whf, Whc, Who};
    uint32_t hi = 0, lo = 0;
#pragma unroll
    for (int p = 0; p < 2; ++p) {
        int k = kt * 32 + (lane >> 4) * 8 + 16 * (j >> 2) + 2 * (j & 3) + p;
        float f = (k < 64) ? Wx[gate][k * 128 + u] : Wh[gate][(k - 64) * 128 + u];
        uint32_t hb = __float_as_uint(f) >> 16;
        float hif = __uint_as_float(hb << 16);
        float lof = f - hif;
        uint32_t lb = __float_as_uint(lof) >> 16;   // truncated residual
        hi |= hb << (16 * p);
        lo |= lb << (16 * p);
    }
    // kt-major tile order inside the plane: tile = kt*8 + nt.
    int inner = ((kt * 8 + nt) * 32 + lane) * 8 + j;
    pack[(size_t)g2 * GRP_STRIDE + inner]          = hi;
    pack[(size_t)g2 * GRP_STRIDE + GRP_DW + inner] = lo;
}

// ---------------------------------------------------------------------------
// Main kernel: 256 threads = 8 wave32s. blockIdx.y = unit group g2 (0..3).
// Wave s computes a 16-row strip x 128 packed cols (4 gates x 32 units).
// Weights (hi+lo, 96 KB) staged once into LDS via TDM, reused across a
// grid-stride loop over row blocks. Epilogue is fully wave-local.
// ---------------------------------------------------------------------------
__global__ void __launch_bounds__(256) lstm_cell_wmma(
    const float* __restrict__ x, const float* __restrict__ h_t,
    const float* __restrict__ c_t,
    const float* __restrict__ b_i, const float* __restrict__ b_f,
    const float* __restrict__ b_c, const float* __restrict__ b_o,
    const uint32_t* __restrict__ pack,
    float* __restrict__ out) {
    extern __shared__ uint32_t lds[]; // 24576 dwords: [0,12288) hi, [12288,24576) lo

    const int tid = threadIdx.x;
    const int g2  = blockIdx.y;
    const uint32_t* gsrc = pack + (size_t)g2 * GRP_STRIDE;

    // Stage this unit-group's 96 KB weight block with the Tensor Data Mover.
    // 1D copy: 12288 x 8-byte elements, contiguous global -> contiguous LDS.
    if (tid < 32) {
        uint64_t ga = (uint64_t)(uintptr_t)gsrc;
        uint32_t la = (uint32_t)(uintptr_t)lds;
        u32x4 d0;
        d0[0] = 1u;                                  // count=1, user descriptor
        d0[1] = la;                                  // lds_addr (bytes)
        d0[2] = (uint32_t)ga;                        // global_addr[31:0]
        d0[3] = (uint32_t)(ga >> 32) | (2u << 30);   // global_addr[56:32] | type=2
        i32x8 d1;
        d1[0] = (int)(3u << 16);                     // data_size = 8 bytes
        d1[1] = (int)(12288u << 16);                 // tensor_dim0[15:0] @bits63:48
        d1[2] = (int)(1u << 16);                     // tensor_dim1 = 1
        d1[3] = (int)(12288u << 16);                 // tile_dim0 @bits127:112
        d1[4] = 0;                                   // tile_dim1/2 unused
        d1[5] = 12288;                               // tensor_dim0_stride
        d1[6] = 0;
        d1[7] = 0;
        i32x4 dz4 = {0, 0, 0, 0};
        i32x8 dz8 = {0, 0, 0, 0, 0, 0, 0, 0};
        __builtin_amdgcn_tensor_load_to_lds(d0, d1, dz4, dz4, dz8, 0);
        __builtin_amdgcn_s_wait_tensorcnt(0);
    }
    __syncthreads();

    const int wave    = tid >> 5;
    const int lane    = tid & 31;
    const int laneLow = lane & 15;
    const int laneHi  = lane >> 4;
    const int uBase   = 32 * g2;

    const uint32_t* ldsHiLane = lds + lane * 8;

    // Per-lane biases (invariant across row blocks).
    float biv[2], bfv[2], bcv[2], bov[2];
#pragma unroll
    for (int half = 0; half < 2; ++half) {
        int u = uBase + 16 * half + laneLow;
        biv[half] = b_i[u];
        bfv[half] = b_f[u];
        bcv[half] = b_c[u];
        bov[half] = b_o[u];
    }

#pragma unroll 1
    for (int rb = blockIdx.x; rb < NRB; rb += gridDim.x) {
        // Compiler-level memory barrier: prevents LICM from hoisting the
        // (loop-invariant) LDS B-tile loads out of this loop, which would
        // blow up register pressure and spill to scratch.
        asm volatile("" ::: "memory");

        const int rowBase = rb * ROWBLK + wave * 16;
        const int arow    = rowBase + laneLow;

        // Prefetch the next row block's A / c_t lines into cache.
        {
            int nrb = rb + gridDim.x;
            if (nrb < NRB) {
                int prow = nrb * ROWBLK + wave * 16 + laneLow;
                __builtin_prefetch(x   + (size_t)prow * 64, 0, 0);
                __builtin_prefetch(h_t + (size_t)prow * 128, 0, 0);
                __builtin_prefetch(c_t + (size_t)prow * 128, 0, 0);
            }
        }

        v8f acc[8];
#pragma unroll
        for (int nt = 0; nt < 8; ++nt) {
            v8f z = {0.f, 0.f, 0.f, 0.f, 0.f, 0.f, 0.f, 0.f};
            acc[nt] = z;
        }

#pragma unroll 1
        for (int kt = 0; kt < KT_N; ++kt) {
            BReg ahi, alo;
            const float* src = (kt < 2)
                ? (x   + (size_t)arow * 64  + kt * 32 + laneHi * 8)
                : (h_t + (size_t)arow * 128 + (kt - 2) * 32 + laneHi * 8);
            cvt_row_bf16x2(src, ahi, alo);
            // Per-kt LDS base (induction variable). Within this iteration:
            //   hi offsets: nt*1024 + {0,16}         <= 7184 bytes
            //   lo offsets: 49152 + nt*1024 + {0,16} <= 56336 bytes
            // -> all fold into 16-bit DS immediates; no VALU address math.
            const uint32_t* pHiK = ldsHiLane + kt * 2048;
#pragma unroll
            for (int nt = 0; nt < 8; ++nt) {
                BReg bhi, blo;
                const uint4* ph = (const uint4*)(pHiK + nt * 256);
                const uint4* pl = (const uint4*)(pHiK + GRP_DW + nt * 256);
                bhi.q[0] = ph[0]; bhi.q[1] = ph[1];
                blo.q[0] = pl[0]; blo.q[1] = pl[1];
                acc[nt] = __builtin_amdgcn_wmma_f32_16x16x32_bf16(
                    false, ahi.v, false, bhi.v, (short)0, acc[nt], false, false);
                acc[nt] = __builtin_amdgcn_wmma_f32_16x16x32_bf16(
                    false, ahi.v, false, blo.v, (short)0, acc[nt], false, false);
                acc[nt] = __builtin_amdgcn_wmma_f32_16x16x32_bf16(
                    false, alo.v, false, bhi.v, (short)0, acc[nt], false, false);
            }
        }

        // Wave-local epilogue: C/D tile layout -> row = r + 8*(lane>>4),
        // col = tile*16 + (lane&15). Gate g lives in acc[2*g + half].
#pragma unroll
        for (int r = 0; r < 8; ++r) {
            const int row = rowBase + r + 8 * laneHi;
#pragma unroll
            for (int half = 0; half < 2; ++half) {
                const int u = uBase + 16 * half + laneLow;
                float ig = fast_sigmoid(acc[0 + half][r] + biv[half]);
                float fg = fast_sigmoid(acc[2 + half][r] + bfv[half]);
                float cg = fast_tanh   (acc[4 + half][r] + bcv[half]);
                float og = fast_sigmoid(acc[6 + half][r] + bov[half]);
                float cold = c_t[(size_t)row * 128 + u];
                float cn = fg * cold + ig * cg;
                float hn = og * fast_tanh(cn);
                out[(size_t)row * 128 + u] = hn;                          // h_new
                out[(size_t)NROWS * 128 + (size_t)row * 128 + u] = cn;    // c_new
            }
        }
    }
}

extern "C" void kernel_launch(void* const* d_in, const int* in_sizes, int n_in,
                              void* d_out, int out_size, void* d_ws, size_t ws_size,
                              hipStream_t stream) {
    const float* x   = (const float*)d_in[0];
    const float* h_t = (const float*)d_in[1];
    const float* c_t = (const float*)d_in[2];
    const float* Wxi = (const float*)d_in[3];
    const float* Whi = (const float*)d_in[4];
    const float* b_i = (const float*)d_in[5];
    const float* Wxf = (const float*)d_in[6];
    const float* Whf = (const float*)d_in[7];
    const float* b_f = (const float*)d_in[8];
    const float* Wxc = (const float*)d_in[9];
    const float* Whc = (const float*)d_in[10];
    const float* b_c = (const float*)d_in[11];
    const float* Wxo = (const float*)d_in[12];
    const float* Who = (const float*)d_in[13];
    const float* b_o = (const float*)d_in[14];

    uint32_t* pack = (uint32_t*)d_ws;   // 4 groups x 96 KB = 384 KB
    float* out = (float*)d_out;

    // 1) Repack weights into WMMA B-operand layout (bf16 hi/lo). 49152 threads.
    hipLaunchKernelGGL(lstm_pack_weights, dim3(192), dim3(256), 0, stream,
                       Wxi, Whi, Wxf, Whf, Wxc, Whc, Wxo, Who, pack);

    // 2) Fused GEMM + LSTM epilogue. grid.y = 4 unit groups; grid-stride over
    //    2048 row blocks with weights resident in 96 KB LDS.
    hipLaunchKernelGGL(lstm_cell_wmma, dim3(256, 4), dim3(256),
                       /*dynamic LDS*/ 24576 * sizeof(uint32_t), stream,
                       x, h_t, c_t, b_i, b_f, b_c, b_o, pack, out);
}
